// VIT_26371099198143
// MI455X (gfx1250) — matrix operand order repaired and checked by reading, not compiled, not run on hardware
//
#include <hip/hip_runtime.h>

// ---------------------------------------------------------------------------
// ViT forward for MI455X (gfx1250, wave32, WMMA). All GEMMs run through one
// batched bf16 WMMA kernel (v_wmma_f32_16x16x32_bf16, f32 accumulate).
// 128x64x32 block tile, 8 waves x (32x32) output, double-buffered LDS with
// global_load_async_to_lds_b128 copies overlapping the WMMA compute.
// ---------------------------------------------------------------------------

typedef __attribute__((ext_vector_type(16))) __bf16 v16bf;
typedef __attribute__((ext_vector_type(8)))  float  v8f;

static __device__ __forceinline__ float bf2f(unsigned short h) {
    return __uint_as_float(((unsigned int)h) << 16);
}
static __device__ __forceinline__ unsigned short f2bf(float f) {
    unsigned int u = __float_as_uint(f);
    u += 0x7fffu + ((u >> 16) & 1u);   // round-to-nearest-even
    return (unsigned short)(u >> 16);
}

// ------------------------------ elementwise --------------------------------

__global__ void cvt_f32_bf16(const float* __restrict__ src,
                             unsigned short* __restrict__ dst, int n) {
    int i = blockIdx.x * 256 + threadIdx.x;
    if (i < n) dst[i] = f2bf(src[i]);
}

// x:(256,3,224,224) -> patches:(256*196, 768) bf16, patch layout c*256+i*16+j
__global__ void patchify(const float* __restrict__ x,
                         unsigned short* __restrict__ out, int total) {
    int i = blockIdx.x * 256 + threadIdx.x;
    if (i >= total) return;
    int q  = i % 768;
    int rp = i / 768;
    int p  = rp % 196;
    int n  = rp / 196;
    int c  = q >> 8;          // /256
    int ij = q & 255;
    int ii = ij >> 4, jj = ij & 15;
    int ph = p / 14, pw = p % 14;
    size_t src = (((size_t)n * 3 + c) * 224 + (size_t)(ph * 16 + ii)) * 224
               + (size_t)(pw * 16 + jj);
    out[i] = f2bf(x[src]);
}

// tok:(256,197,192) bf16. Row 0 <- cls+pos[0]; rows 1..196 += pos[t].
__global__ void add_cls_pos(unsigned short* __restrict__ tok,
                            const float* __restrict__ cls,
                            const float* __restrict__ pos, int total) {
    int i = blockIdx.x * 256 + threadIdx.x;
    if (i >= total) return;
    int d = i % 192;
    int t = (i / 192) % 197;
    float v;
    if (t == 0) v = cls[d] + pos[d];
    else        v = bf2f(tok[i]) + pos[t * 192 + d];
    tok[i] = f2bf(v);
}

// One wave (32 lanes) per softmax row of length len, row stride ld (bf16).
__global__ void softmax_rows(unsigned short* __restrict__ s,
                             int rows, int len, int ld) {
    int wave = threadIdx.x >> 5;
    int lane = threadIdx.x & 31;
    int row  = blockIdx.x * 8 + wave;
    if (row >= rows) return;
    unsigned short* p = s + (size_t)row * ld;
    float mx = -3.0e38f;
    for (int j = lane; j < len; j += 32) mx = fmaxf(mx, bf2f(p[j]));
    #pragma unroll
    for (int o = 16; o > 0; o >>= 1) mx = fmaxf(mx, __shfl_xor(mx, o, 32));
    float sum = 0.f;
    for (int j = lane; j < len; j += 32) sum += __expf(bf2f(p[j]) - mx);
    #pragma unroll
    for (int o = 16; o > 0; o >>= 1) sum += __shfl_xor(sum, o, 32);
    float inv = 1.f / sum;
    for (int j = lane; j < len; j += 32)
        p[j] = f2bf(__expf(bf2f(p[j]) - mx) * inv);
}

// ------------------------------ WMMA GEMM ----------------------------------
// C[z] = A[z] (MxK) * op(B[z]) (+bias)(+resid)(relu)(scale), bf16 in, f32 acc.
// TRANS_B=true : B is (N x K) row-major (PyTorch weight, and Q·K^T scores)
// TRANS_B=false: B is (K x N) row-major (probs · V)
// Batch offsets: off(z) = (z/inner)*Out + (z%inner)*In   (element counts)
// flags: 1 = relu, 2 = f32 output (else bf16)

#define WMMA_BF16(A_, B_, C_) \
    __builtin_amdgcn_wmma_f32_16x16x32_bf16(false, (A_), false, (B_), (short)0, (C_), false, false)

template <bool TRANS_B>
__global__ void __launch_bounds__(256)
gemm_wmma(const unsigned short* __restrict__ A,
          const unsigned short* __restrict__ Bm,
          void* __restrict__ Cv,
          const unsigned short* __restrict__ bias,
          const unsigned short* __restrict__ resid,
          int M, int N, int K, int lda, int ldb, int ldc,
          long long aO, long long aI, long long bO, long long bI,
          long long cO, long long cI, int inner,
          float scale, int flags) {
    constexpr int LDA_S = 40;                 // 32 + 8 pad (halves)
    constexpr int LDB_S = TRANS_B ? 40 : 72;  // (64x32)+pad or (32x64)+pad
    constexpr int BSZ   = TRANS_B ? 64 * 40 : 32 * 72;
    __shared__ unsigned short As[2][128 * 40];
    __shared__ unsigned short Bs[2][BSZ];

    const int tid  = threadIdx.x;
    const int lane = tid & 31;
    const int wave = tid >> 5;
    const int wm   = wave >> 1;     // 0..3 : 32-row strip
    const int wn   = wave & 1;      // 0..1 : 32-col strip
    const int sel  = lane >> 4;     // lane-half select
    const int lidx = lane & 15;

    const long long z  = blockIdx.z;
    const long long zo = z / inner, zi = z % inner;
    const unsigned short* Ab = A  + zo * aO + zi * aI;
    const unsigned short* Bb = Bm + zo * bO + zi * bI;
    const long long cOff = zo * cO + zi * cI;

    const int m0 = blockIdx.y * 128;
    const int n0 = blockIdx.x * 64;

    const bool alA = ((((unsigned long long)Ab) & 15ull) == 0) && ((lda & 7) == 0);
    const bool alB = ((((unsigned long long)Bb) & 15ull) == 0) && ((ldb & 7) == 0);

    // Stage tile kt into LDS buffer `buf` (async fast path / clamped tail).
    auto stage = [&](int kt, int buf) {
        const int k0 = kt << 5;
        // ---- A: 128 rows x 32 cols (512 16B-chunks, 2 per thread) ----
        #pragma unroll
        for (int it = 0; it < 2; ++it) {
            int c = tid + it * 256;
            int r = c >> 2, cch = (c & 3) << 3;
            unsigned short* d = &As[buf][r * LDA_S + cch];
            int gr = m0 + r, gc = k0 + cch;
            if (alA && gr < M && gc + 8 <= K) {
                // CDNA5 async copy: global -> LDS, tracked on ASYNCcnt
                unsigned ldsOff = (unsigned)(unsigned long long)(void*)d;
                asm volatile("global_load_async_to_lds_b128 %0, %1, off"
                             :: "v"(ldsOff),
                                "v"((unsigned long long)(Ab + (size_t)gr * lda + gc))
                             : "memory");
                __builtin_prefetch(Ab + (size_t)gr * lda + gc + 32, 0, 1);
            } else {
                int grc = gr < M - 1 ? gr : M - 1;      // clamp: branch-free OOB
                #pragma unroll
                for (int e = 0; e < 8; ++e) {
                    int gcc = gc + e < K - 1 ? gc + e : K - 1;
                    unsigned short v = Ab[(size_t)grc * lda + gcc];
                    d[e] = (gr < M && gc + e < K) ? v : (unsigned short)0;
                }
            }
        }
        // ---- B ----
        if (TRANS_B) { // 64 rows(N) x 32 cols(K)
            int r = tid >> 2, cch = (tid & 3) << 3;
            unsigned short* d = &Bs[buf][r * LDB_S + cch];
            int gr = n0 + r, gc = k0 + cch;
            if (alB && gr < N && gc + 8 <= K) {
                unsigned ldsOff = (unsigned)(unsigned long long)(void*)d;
                asm volatile("global_load_async_to_lds_b128 %0, %1, off"
                             :: "v"(ldsOff),
                                "v"((unsigned long long)(Bb + (size_t)gr * ldb + gc))
                             : "memory");
            } else {
                int grc = gr < N - 1 ? gr : N - 1;
                #pragma unroll
                for (int e = 0; e < 8; ++e) {
                    int gcc = gc + e < K - 1 ? gc + e : K - 1;
                    unsigned short v = Bb[(size_t)grc * ldb + gcc];
                    d[e] = (gr < N && gc + e < K) ? v : (unsigned short)0;
                }
            }
        } else {       // 32 rows(K) x 64 cols(N)
            int r = tid >> 3, cch = (tid & 7) << 3;
            unsigned short* d = &Bs[buf][r * LDB_S + cch];
            int gr = k0 + r, gc = n0 + cch;
            if (alB && gr < K && gc + 8 <= N) {
                unsigned ldsOff = (unsigned)(unsigned long long)(void*)d;
                asm volatile("global_load_async_to_lds_b128 %0, %1, off"
                             :: "v"(ldsOff),
                                "v"((unsigned long long)(Bb + (size_t)gr * ldb + gc))
                             : "memory");
            } else {
                int grc = gr < K - 1 ? gr : K - 1;
                #pragma unroll
                for (int e = 0; e < 8; ++e) {
                    int gcc = gc + e < N - 1 ? gc + e : N - 1;
                    unsigned short v = Bb[(size_t)grc * ldb + gcc];
                    d[e] = (gr < K && gc + e < N) ? v : (unsigned short)0;
                }
            }
        }
    };

    v8f acc00 = {}, acc01 = {}, acc10 = {}, acc11 = {};

    const int ksteps = (K + 31) >> 5;
    stage(0, 0);                               // prologue: async-fill buffer 0
    for (int kt = 0; kt < ksteps; ++kt) {
        const int cur = kt & 1;
        // Tile kt landed for this wave, then make it workgroup-visible.
        asm volatile("s_wait_asynccnt 0x0" ::: "memory");
        __syncthreads();
        // Issue async copies for tile kt+1 -> other buffer; they overlap with
        // the WMMAs below. (Safe: the barrier above proved every wave finished
        // reading that buffer during iteration kt-1.)
        if (kt + 1 < ksteps) stage(kt + 1, cur ^ 1);

        // ---- fragments per ISA 16-bit layouts (wave32) ----
        union Frag { v16bf v; unsigned int u[8]; };
        Frag a0, a1, b0, b1;
        #pragma unroll
        for (int r = 0; r < 8; ++r) { // A: K = 2r + (r>=4?8:0) + sel*8
            int kk = 2 * r + ((r >= 4) ? 8 : 0) + sel * 8;
            a0.u[r] = *(const unsigned int*)&As[cur][(wm * 32 + lidx) * LDA_S + kk];
            a1.u[r] = *(const unsigned int*)&As[cur][(wm * 32 + 16 + lidx) * LDA_S + kk];
        }
        if (TRANS_B) {
            #pragma unroll
            for (int r = 0; r < 8; ++r) { // B: K = 2r + (r>=4?8:0) + sel*8, N = lidx
                int kk = 2 * r + ((r >= 4) ? 8 : 0) + sel * 8;
                b0.u[r] = *(const unsigned int*)&Bs[cur][(wn * 32 + lidx) * LDB_S + kk];
                b1.u[r] = *(const unsigned int*)&Bs[cur][(wn * 32 + 16 + lidx) * LDB_S + kk];
            }
        } else {
            #pragma unroll
            for (int r = 0; r < 8; ++r) {
                int kk = 2 * r + ((r >= 4) ? 8 : 0) + sel * 8;
                unsigned int lo0 = Bs[cur][kk * LDB_S + wn * 32 + lidx];
                unsigned int hi0 = Bs[cur][(kk + 1) * LDB_S + wn * 32 + lidx];
                b0.u[r] = lo0 | (hi0 << 16);
                unsigned int lo1 = Bs[cur][kk * LDB_S + wn * 32 + 16 + lidx];
                unsigned int hi1 = Bs[cur][(kk + 1) * LDB_S + wn * 32 + 16 + lidx];
                b1.u[r] = lo1 | (hi1 << 16);
            }
        }
        acc00 = WMMA_BF16(a0.v, b0.v, acc00);
        acc01 = WMMA_BF16(a0.v, b1.v, acc01);
        acc10 = WMMA_BF16(a1.v, b0.v, acc10);
        acc11 = WMMA_BF16(a1.v, b1.v, acc11);
    }

    // ---- epilogue: scale, bias, residual, relu, store (bf16 or f32) ----
    const bool relu = (flags & 1) != 0;
    const bool f32o = (flags & 2) != 0;
    float* Cf = (float*)Cv;
    unsigned short* Ch = (unsigned short*)Cv;
    #pragma unroll
    for (int ms = 0; ms < 2; ++ms) {
        #pragma unroll
        for (int ns = 0; ns < 2; ++ns) {
            v8f acc = ms == 0 ? (ns == 0 ? acc00 : acc01)
                              : (ns == 0 ? acc10 : acc11);
            int gn = n0 + wn * 32 + ns * 16 + lidx;
            #pragma unroll
            for (int r = 0; r < 8; ++r) {
                int gm = m0 + wm * 32 + ms * 16 + r + sel * 8; // C/D: M=r+sel*8
                if (gm < M && gn < N) {
                    float v = acc[r] * scale;
                    if (bias)  v += bf2f(bias[gn]);
                    size_t ci = (size_t)cOff + (size_t)gm * ldc + gn;
                    if (resid) v += bf2f(resid[ci]);
                    if (relu)  v = fmaxf(v, 0.f);
                    if (f32o) Cf[ci] = v; else Ch[ci] = f2bf(v);
                }
            }
        }
    }
}

// ------------------------------- launcher ----------------------------------

extern "C" void kernel_launch(void* const* d_in, const int* in_sizes, int n_in,
                              void* d_out, int out_size, void* d_ws, size_t ws_size,
                              hipStream_t stream) {
    constexpr int Bsz = 256, Ttok = 197, D = 192, NH = 4, DH = 48;
    constexpr int NBLK = 3, OUTN = 1000, NPAT = 196, PK = 768, SLD = 208;
    constexpr int MT = Bsz * Ttok;                       // 50432 token rows
    (void)n_in; (void)out_size; (void)ws_size;

    char* ws = (char*)d_ws;
    size_t off = 0;
    auto alloc = [&](size_t elems) -> unsigned short* {
        unsigned short* p = (unsigned short*)(ws + off);
        off += ((elems * 2 + 255) & ~(size_t)255);
        return p;
    };

    // bf16 copies of all weights/biases (inputs 1..20)
    unsigned short* wbf[21];
    for (int i = 1; i <= 20; ++i) wbf[i] = alloc((size_t)in_sizes[i]);

    unsigned short* tok    = alloc((size_t)MT * D);
    unsigned short* tq     = alloc((size_t)MT * D);
    unsigned short* tk     = alloc((size_t)MT * D);
    unsigned short* tv     = alloc((size_t)MT * D);
    unsigned short* tqh    = alloc((size_t)MT * D);
    unsigned short* tkh    = alloc((size_t)MT * D);
    unsigned short* tvh    = alloc((size_t)MT * D);
    unsigned short* to_    = alloc((size_t)MT * D);
    unsigned short* scores = alloc((size_t)Bsz * NH * Ttok * SLD);
    unsigned short* hidden = alloc((size_t)MT * 4 * D);  // also patches buffer
    unsigned short* patches = hidden;                    // disjoint live range

    // Convert weights to bf16
    for (int i = 1; i <= 20; ++i) {
        int n = in_sizes[i];
        cvt_f32_bf16<<<dim3((n + 255) / 256), dim3(256), 0, stream>>>(
            (const float*)d_in[i], wbf[i], n);
    }

    auto gemm = [&](bool transB, const unsigned short* A, const unsigned short* W,
                    void* C, const unsigned short* bias, const unsigned short* resid,
                    int M, int N, int K, int lda, int ldb, int ldc,
                    long long aO, long long aI, long long bO, long long bI,
                    long long cO, long long cI, int inner, int nbatch,
                    float scale, int flags) {
        dim3 grid((N + 63) / 64, (M + 127) / 128, nbatch), blk(256);
        if (transB)
            gemm_wmma<true><<<grid, blk, 0, stream>>>(A, W, C, bias, resid,
                M, N, K, lda, ldb, ldc, aO, aI, bO, bI, cO, cI, inner, scale, flags);
        else
            gemm_wmma<false><<<grid, blk, 0, stream>>>(A, W, C, bias, resid,
                M, N, K, lda, ldb, ldc, aO, aI, bO, bI, cO, cI, inner, scale, flags);
    };

    // 1) Patch embed: patches (b:196x768) @ lm_w^T -> tok rows 1..196
    {
        int tot = Bsz * NPAT * PK;
        patchify<<<dim3((tot + 255) / 256), dim3(256), 0, stream>>>(
            (const float*)d_in[0], patches, tot);
        gemm(true, patches, wbf[1], tok + D, wbf[2], nullptr,
             NPAT, D, PK, PK, PK, D,
             (long long)NPAT * PK, 0, 0, 0, (long long)Ttok * D, 0,
             1, Bsz, 1.f, 0);
        int tot2 = MT * D;
        add_cls_pos<<<dim3((tot2 + 255) / 256), dim3(256), 0, stream>>>(
            tok, (const float*)d_in[3], (const float*)d_in[4], tot2);
    }

    // 2) Transformer blocks
    for (int ib = 0; ib < NBLK; ++ib) {
        const unsigned short* qw  = wbf[5]  + (size_t)ib * D * D;
        const unsigned short* qbs = wbf[6]  + (size_t)ib * D;
        const unsigned short* kw  = wbf[7]  + (size_t)ib * D * D;
        const unsigned short* kbs = wbf[8]  + (size_t)ib * D;
        const unsigned short* vw  = wbf[9]  + (size_t)ib * D * D;
        const unsigned short* vbs = wbf[10] + (size_t)ib * D;
        const unsigned short* inw = wbf[11] + (size_t)ib * 3 * D * D;
        const unsigned short* inb = wbf[12] + (size_t)ib * 3 * D;
        const unsigned short* ow  = wbf[13] + (size_t)ib * D * D;
        const unsigned short* obs = wbf[14] + (size_t)ib * D;
        const unsigned short* uw  = wbf[15] + (size_t)ib * 4 * D * D;
        const unsigned short* ub  = wbf[16] + (size_t)ib * 4 * D;
        const unsigned short* dw  = wbf[17] + (size_t)ib * D * 4 * D;
        const unsigned short* db  = wbf[18] + (size_t)ib * D;

        gemm(true, tok, qw, tq, qbs, nullptr, MT, D, D, D, D, D,
             0, 0, 0, 0, 0, 0, 1, 1, 1.f, 0);
        gemm(true, tok, kw, tk, kbs, nullptr, MT, D, D, D, D, D,
             0, 0, 0, 0, 0, 0, 1, 1, 1.f, 0);
        gemm(true, tok, vw, tv, vbs, nullptr, MT, D, D, D, D, D,
             0, 0, 0, 0, 0, 0, 1, 1, 1.f, 0);
        gemm(true, tq, inw,             tqh, inb,         nullptr, MT, D, D, D, D, D,
             0, 0, 0, 0, 0, 0, 1, 1, 1.f, 0);
        gemm(true, tk, inw + D * D,     tkh, inb + D,     nullptr, MT, D, D, D, D, D,
             0, 0, 0, 0, 0, 0, 1, 1, 1.f, 0);
        gemm(true, tv, inw + 2 * D * D, tvh, inb + 2 * D, nullptr, MT, D, D, D, D, D,
             0, 0, 0, 0, 0, 0, 1, 1, 1.f, 0);

        // scores[b,h] = (qh · kh^T) / sqrt(dh)   (batch z = b*NH + h)
        gemm(true, tqh, tkh, scores, nullptr, nullptr,
             Ttok, Ttok, DH, D, D, SLD,
             (long long)Ttok * D, DH, (long long)Ttok * D, DH,
             (long long)NH * Ttok * SLD, (long long)Ttok * SLD,
             NH, Bsz * NH, 0.14433756729740643f, 0);

        int rows = Bsz * NH * Ttok;
        softmax_rows<<<dim3((rows + 7) / 8), dim3(256), 0, stream>>>(
            scores, rows, Ttok, SLD);

        // o[b,h] = probs · vh   (non-transposed B)
        gemm(false, scores, tvh, to_, nullptr, nullptr,
             Ttok, DH, Ttok, SLD, D, D,
             (long long)NH * Ttok * SLD, (long long)Ttok * SLD,
             (long long)Ttok * D, DH,
             (long long)Ttok * D, DH,
             NH, Bsz * NH, 1.f, 0);

        // attention out-proj with residual (tok = tok + o·W^T + b)
        gemm(true, to_, ow, tok, obs, tok, MT, D, D, D, D, D,
             0, 0, 0, 0, 0, 0, 1, 1, 1.f, 0);
        // MLP: tok = (relu(tok·up^T + ub)) · down^T + db   (no residual)
        gemm(true, tok, uw, hidden, ub, nullptr, MT, 4 * D, D, D, D, 4 * D,
             0, 0, 0, 0, 0, 0, 1, 1, 1.f, 1 /*relu*/);
        gemm(true, hidden, dw, tok, db, nullptr, MT, D, 4 * D, 4 * D, 4 * D, D,
             0, 0, 0, 0, 0, 0, 1, 1, 1.f, 0);
    }

    // 3) Head on cls rows (row stride Ttok*D picks t==0), f32 out
    gemm(true, tok, wbf[19], d_out, wbf[20], nullptr,
         Bsz, OUTN, D, Ttok * D, D, OUTN,
         0, 0, 0, 0, 0, 0, 1, 1, 1.f, 2 /*f32 out*/);
}
